// SPIL_Layer_45363444580521
// MI455X (gfx1250) — matrix operand order repaired and checked by reading, not compiled, not run on hardware
//
#include <hip/hip_runtime.h>
#include <stdint.h>

// Problem constants (fixed by the reference).
#define BB 8
#define NN 4096
#define KK 32
#define CC 128
#define HD 64
#define NPTS (BB*NN)        // 32768 points
#define D_TH 0.04f

typedef float    v8f   __attribute__((ext_vector_type(8)));
typedef float    v4f   __attribute__((ext_vector_type(4)));
typedef __bf16   v16bf __attribute__((ext_vector_type(16)));
typedef unsigned v4u   __attribute__((ext_vector_type(4)));
typedef unsigned u32x4 __attribute__((ext_vector_type(4)));
typedef int      v8i   __attribute__((ext_vector_type(8)));
typedef int      v4i   __attribute__((ext_vector_type(4)));

union BFrag { v16bf v; v4u u[2]; };

static __device__ __forceinline__ v16bf pack16(v4f a, v4f b, v4f c, v4f d) {
  v16bf r;
#pragma unroll
  for (int i = 0; i < 4; ++i) {
    r[i]      = (__bf16)a[i];
    r[4 + i]  = (__bf16)b[i];
    r[8 + i]  = (__bf16)c[i];
    r[12 + i] = (__bf16)d[i];
  }
  return r;
}

// A-fragment (16x32 bf16) for v_wmma_f32_16x16x32_bf16 from a row-major fp32
// matrix with leading dim ld. Lanes 0-15 hold row=lane,
// K = c0+0..15; lanes 16-31 hold row=lane-16, K = c0+16..31.
static __device__ __forceinline__ v16bf afrag(const float* M, int row0, int c0,
                                              int ld, int lane) {
  int hf = lane >> 4, l = lane & 15;
  const float* p = M + (size_t)(row0 + l) * ld + c0 + hf * 16;
  v4f x0 = *(const v4f*)(p);
  v4f x1 = *(const v4f*)(p + 4);
  v4f x2 = *(const v4f*)(p + 8);
  v4f x3 = *(const v4f*)(p + 12);
  return pack16(x0, x1, x2, x3);
}

// B-fragment (32x16 bf16) from row-major (K x N) fp32 weights, ldw = N.
// Lanes 0-15: col = n0+lane, K = c0+0..15; lanes 16-31: col = n0+lane-16,
// K = c0+16..31 (ascending elements).
static __device__ __forceinline__ v16bf bfrag(const float* W, int c0, int n0,
                                              int ldw, int lane) {
  int hf = lane >> 4, l = lane & 15;
  const float* p = W + (size_t)(c0 + hf * 16) * ldw + n0 + l;
  v16bf r;
#pragma unroll
  for (int i = 0; i < 16; ++i) r[i] = (__bf16)p[(size_t)i * ldw];
  return r;
}

static __device__ __forceinline__ v8f wmma_bf16(v16bf a, v16bf b, v8f c) {
  return __builtin_amdgcn_wmma_f32_16x16x32_bf16(false, a, false, b,
                                                 (short)0, c, false, false);
}

// ---------------------------------------------------------------------------
// Kernel 1: FI = relu(CF @ phi_w + phi_b)  (WMMA, wave = 16 points)
//           AI = dot(relu(cxyz @ m1_w + m1_b), psi_w[0:32])
// ---------------------------------------------------------------------------
__global__ __launch_bounds__(128) void k_prep(
    const float* __restrict__ cxyz, const float* __restrict__ cf,
    const float* __restrict__ phi_w, const float* __restrict__ phi_b,
    const float* __restrict__ m1_w, const float* __restrict__ m1_b,
    const float* __restrict__ psi_w,
    float* __restrict__ FI, float* __restrict__ AI) {
  int lane = threadIdx.x & 31;
  int wave = (blockIdx.x * blockDim.x + threadIdx.x) >> 5;
  int p0 = wave * 16;
  if (p0 >= NPTS) return;

  v8f acc[4] = {};
#pragma unroll
  for (int kk = 0; kk < 4; ++kk) {
    v16bf a = afrag(cf, p0, kk * 32, CC, lane);
#pragma unroll
    for (int nt = 0; nt < 4; ++nt) {
      v16bf b = bfrag(phi_w, kk * 32, nt * 16, HD, lane);
      acc[nt] = wmma_bf16(a, b, acc[nt]);
    }
  }
  int hf = lane >> 4, cl = lane & 15;
#pragma unroll
  for (int nt = 0; nt < 4; ++nt) {
    int col = nt * 16 + cl;
    float bias = phi_b[col];
#pragma unroll
    for (int v = 0; v < 8; ++v) {
      int row = p0 + v + 8 * hf;              // C/D layout: vgpr v, half hf
      FI[(size_t)row * HD + col] = fmaxf(acc[nt][v] + bias, 0.f);
    }
  }
  if (lane < 16) {                             // 16 points per wave
    int pt = p0 + lane;
    float x = cxyz[pt * 3 + 0], y = cxyz[pt * 3 + 1], z = cxyz[pt * 3 + 2];
    float a = 0.f;
#pragma unroll 8
    for (int p = 0; p < 32; ++p) {
      float t = fmaf(x, m1_w[p], fmaf(y, m1_w[32 + p], fmaf(z, m1_w[64 + p], m1_b[p])));
      a = fmaf(fmaxf(t, 0.f), psi_w[p], a);
    }
    AI[pt] = a;
  }
}

// ---------------------------------------------------------------------------
// TDM: DMA one contiguous 16KB NF tile (32x128 fp32) into LDS.
// D# per CDNA5 ISA: 1-D tile, data_size=4B, type=2.
// clang-23 toolchain: 6-arg form (g0, g1, g2, g3, g4, cpol).
// ---------------------------------------------------------------------------
static __device__ __forceinline__ void tdm_load_tile(const float* gptr,
                                                     unsigned lds_off) {
  unsigned long long ga = (unsigned long long)(uintptr_t)gptr;
  u32x4 g0;
  g0[0] = 1u;                                        // count=1, user descriptor
  g0[1] = lds_off;                                   // lds_addr
  g0[2] = (unsigned)(ga & 0xffffffffull);            // global_addr[31:0]
  g0[3] = (unsigned)((ga >> 32) & 0x01ffffffull) | (2u << 30); // addr hi | type=2
  v8i g1;
  g1[0] = (int)(2u << 16);                           // data_size = 4 bytes
  g1[1] = (int)((unsigned)(KK * CC) << 16);          // tensor_dim0[15:0]
  g1[2] = (int)(1u << 16);                           // tensor_dim0 hi | tensor_dim1=1
  g1[3] = (int)((unsigned)(KK * CC) << 16);          // tensor_dim1 hi | tile_dim0=4096
  g1[4] = 1;                                         // tile_dim1=1, tile_dim2=0
  g1[5] = KK * CC;                                   // tensor_dim0_stride lo
  g1[6] = 0;
  g1[7] = 0;
  v4i z4 = {0, 0, 0, 0};
  v8i z8 = {0, 0, 0, 0, 0, 0, 0, 0};
  asm volatile("" ::: "memory");                     // keep prior LDS use ordered
  __builtin_amdgcn_tensor_load_to_lds(g0, g1, z4, z4, z8, 0);
}

// ---------------------------------------------------------------------------
// Kernel 2: the streamer. Per point: theta-GEMM (WMMA bf16), r_f reduction,
// positional score + SPIL mask, masked weighted softmax over K=32 lanes,
// u = w^T NF, emit u (128) and s = sum(W).
// ---------------------------------------------------------------------------
#define WPB 4     // waves per block
#define PPW 8     // points per wave

__global__ __launch_bounds__(128) void k_main(
    const float* __restrict__ cxyz, const float* __restrict__ nxyz,
    const float* __restrict__ nf,
    const float* __restrict__ theta_w, const float* __restrict__ theta_b,
    const float* __restrict__ m2_w, const float* __restrict__ m2_b,
    const float* __restrict__ psi_w, const float* __restrict__ psi_b,
    const float* __restrict__ FI, const float* __restrict__ AI,
    float* __restrict__ U, float* __restrict__ S) {
  __shared__ float s_nf[WPB][2][KK * CC];   // 128 KB: double-buffered NF tiles
  __shared__ v4u   s_th[16][32][2];         // 16 KB: pre-swizzled theta bf16 B-frags
  __shared__ float s_tb[HD];
  __shared__ v4f   s_m2[32];                // {m2x, m2y, m2z, m2b}
  __shared__ float s_p2[32];                // psi_w[32:64]
  __shared__ float s_rf[WPB][KK];

  int lane = threadIdx.x & 31;
  int wv   = threadIdx.x >> 5;
  int hf = lane >> 4, cl = lane & 15;

  // Stage swizzled theta_w (each wave builds 4 of the 16 (kk,nt) tiles).
#pragma unroll
  for (int t = 0; t < 4; ++t) {
    int tile = wv * 4 + t;
    int kk = tile >> 2, nt = tile & 3;
    BFrag fr;
    fr.v = bfrag(theta_w, kk * 32, nt * 16, HD, lane);
    s_th[tile][lane][0] = fr.u[0];
    s_th[tile][lane][1] = fr.u[1];
  }
  for (int i = threadIdx.x; i < HD; i += blockDim.x) s_tb[i] = theta_b[i];
  if (threadIdx.x < 32) {
    int p = threadIdx.x;
    v4f m;
    m.x = m2_w[p]; m.y = m2_w[32 + p]; m.z = m2_w[64 + p]; m.w = m2_b[p];
    s_m2[p] = m;
    s_p2[p] = psi_w[32 + p];
  }
  __syncthreads();

  int wave  = blockIdx.x * WPB + wv;
  int pbase = wave * PPW;
  float psib = psi_b[0];

  float* buf0 = s_nf[wv][0];
  float* buf1 = s_nf[wv][1];
  unsigned off0 = (unsigned)(uintptr_t)buf0;   // generic->LDS offset (low 32b)
  unsigned off1 = (unsigned)(uintptr_t)buf1;

  tdm_load_tile(nf + (size_t)pbase * (KK * CC), off0);

  for (int it = 0; it < PPW; ++it) {
    int n = pbase + it;
    const float* cur = (it & 1) ? buf1 : buf0;
    if (it + 1 < PPW) {
      tdm_load_tile(nf + (size_t)(n + 1) * (KK * CC), ((it + 1) & 1) ? off1 : off0);
      __builtin_amdgcn_s_wait_tensorcnt(1);    // oldest (cur) complete
    } else {
      __builtin_amdgcn_s_wait_tensorcnt(0);
    }
    asm volatile("" ::: "memory");

    // ---- FJ = relu(NF @ theta_w + theta_b), fused into r_f partials ----
    v8f acc[2][4] = {};
#pragma unroll
    for (int kk = 0; kk < 4; ++kk) {
      v16bf a0 = afrag(cur, 0,  kk * 32, CC, lane);
      v16bf a1 = afrag(cur, 16, kk * 32, CC, lane);
#pragma unroll
      for (int nt = 0; nt < 4; ++nt) {
        BFrag fr;
        fr.u[0] = s_th[kk * 4 + nt][lane][0];
        fr.u[1] = s_th[kk * 4 + nt][lane][1];
        acc[0][nt] = wmma_bf16(a0, fr.v, acc[0][nt]);
        acc[1][nt] = wmma_bf16(a1, fr.v, acc[1][nt]);
      }
    }

    float rfp[2][8];
#pragma unroll
    for (int mt = 0; mt < 2; ++mt)
#pragma unroll
      for (int v = 0; v < 8; ++v) rfp[mt][v] = 0.f;

#pragma unroll
    for (int nt = 0; nt < 4; ++nt) {
      int col = nt * 16 + cl;
      float fi = FI[(size_t)n * HD + col];
      float tb = s_tb[col];
#pragma unroll
      for (int mt = 0; mt < 2; ++mt)
#pragma unroll
        for (int v = 0; v < 8; ++v)
          rfp[mt][v] = fmaf(fmaxf(acc[mt][nt][v] + tb, 0.f), fi, rfp[mt][v]);
    }
    // Row-sum across the 16 lanes of each half (masks < 16 stay in-half).
#pragma unroll
    for (int mt = 0; mt < 2; ++mt)
#pragma unroll
      for (int v = 0; v < 8; ++v) {
        float s = rfp[mt][v];
        s += __shfl_xor(s, 1, 32);
        s += __shfl_xor(s, 2, 32);
        s += __shfl_xor(s, 4, 32);
        s += __shfl_xor(s, 8, 32);
        rfp[mt][v] = s;
      }
    if (cl == 0) {                              // lanes 0 and 16 publish
#pragma unroll
      for (int v = 0; v < 8; ++v) {
        s_rf[wv][hf * 8 + v]      = rfp[0][v];  // rows mt0: hf*8+v
        s_rf[wv][16 + hf * 8 + v] = rfp[1][v];  // rows mt1
      }
    }
    asm volatile("s_wait_dscnt 0" ::: "memory"); // wave-local LDS handoff
    float rf = s_rf[wv][lane] * 0.125f;          // /sqrt(HEAD_DIM)

    // ---- positional score + SPIL mask (lane == neighbor k) ----
    const float* nx = nxyz + ((size_t)n * KK + lane) * 3;
    float x = nx[0], y = nx[1], z = nx[2];
    float cx = cxyz[n * 3 + 0], cy = cxyz[n * 3 + 1], cz = cxyz[n * 3 + 2];
    float rl = 0.f;
#pragma unroll 8
    for (int p = 0; p < 32; ++p) {
      v4f m = s_m2[p];
      float t = fmaf(x, m.x, fmaf(y, m.y, fmaf(z, m.z, m.w)));
      rl = fmaf(fmaxf(t, 0.f), s_p2[p], rl);
    }
    float rlb = fmaxf(AI[n] + rl + psib, 0.f);
    float dx = cx - x, dy = cy - y, dz = cz - z;
    float dist = sqrtf(dx * dx + dy * dy + dz * dz);
    if ((cz == z) && (dist > D_TH)) rlb = 0.f;   // same-frame & far => masked

    // ---- masked, r_l-weighted softmax over the 32 lanes ----
    float mx = rf;
    mx = fmaxf(mx, __shfl_xor(mx, 16, 32));
    mx = fmaxf(mx, __shfl_xor(mx, 8, 32));
    mx = fmaxf(mx, __shfl_xor(mx, 4, 32));
    mx = fmaxf(mx, __shfl_xor(mx, 2, 32));
    mx = fmaxf(mx, __shfl_xor(mx, 1, 32));
    float num = rlb * __expf(rf - mx);
    float sum = num;
    sum += __shfl_xor(sum, 16, 32);
    sum += __shfl_xor(sum, 8, 32);
    sum += __shfl_xor(sum, 4, 32);
    sum += __shfl_xor(sum, 2, 32);
    sum += __shfl_xor(sum, 1, 32);
    float inv = 1.f / (sum + 1e-8f);
    float w = num * inv;

    // ---- u = w^T NF (lane owns 4 feature columns), straight from LDS ----
    v4f u = {0.f, 0.f, 0.f, 0.f};
    const float* base = cur + lane * 4;
#pragma unroll 8
    for (int k = 0; k < KK; ++k) {
      float wk = __shfl(w, k, 32);
      v4f row = *(const v4f*)(base + (size_t)k * CC);
      u.x = fmaf(wk, row.x, u.x);
      u.y = fmaf(wk, row.y, u.y);
      u.z = fmaf(wk, row.z, u.z);
      u.w = fmaf(wk, row.w, u.w);
    }
    *(v4f*)(U + (size_t)n * CC + lane * 4) = u;
    if (lane == 0) S[n] = sum * inv;             // sum(W_ij)
  }
}

// ---------------------------------------------------------------------------
// Kernel 3: OUT = U @ z_w + s * z_b  (WMMA, wave = 16 points)
// ---------------------------------------------------------------------------
__global__ __launch_bounds__(128) void k_out(
    const float* __restrict__ U, const float* __restrict__ S,
    const float* __restrict__ z_w, const float* __restrict__ z_b,
    float* __restrict__ out) {
  int lane = threadIdx.x & 31;
  int wave = (blockIdx.x * blockDim.x + threadIdx.x) >> 5;
  int p0 = wave * 16;
  if (p0 >= NPTS) return;

  v8f acc[4] = {};
#pragma unroll
  for (int kk = 0; kk < 4; ++kk) {
    v16bf a = afrag(U, p0, kk * 32, CC, lane);
#pragma unroll
    for (int nt = 0; nt < 4; ++nt) {
      v16bf b = bfrag(z_w, kk * 32, nt * 16, HD, lane);
      acc[nt] = wmma_bf16(a, b, acc[nt]);
    }
  }
  int hf = lane >> 4, cl = lane & 15;
  float sv[8];
#pragma unroll
  for (int v = 0; v < 8; ++v) sv[v] = S[p0 + v + 8 * hf];
#pragma unroll
  for (int nt = 0; nt < 4; ++nt) {
    int col = nt * 16 + cl;
    float zb = z_b[col];
#pragma unroll
    for (int v = 0; v < 8; ++v) {
      int row = p0 + v + 8 * hf;
      out[(size_t)row * HD + col] = fmaf(sv[v], zb, acc[nt][v]);
    }
  }
}

// ---------------------------------------------------------------------------
extern "C" void kernel_launch(void* const* d_in, const int* in_sizes, int n_in,
                              void* d_out, int out_size, void* d_ws, size_t ws_size,
                              hipStream_t stream) {
  const float* cxyz    = (const float*)d_in[0];
  const float* cf      = (const float*)d_in[1];
  const float* nxyz    = (const float*)d_in[2];
  const float* nf      = (const float*)d_in[3];
  const float* phi_w   = (const float*)d_in[4];
  const float* phi_b   = (const float*)d_in[5];
  const float* theta_w = (const float*)d_in[6];
  const float* theta_b = (const float*)d_in[7];
  const float* m1_w    = (const float*)d_in[8];
  const float* m1_b    = (const float*)d_in[9];
  const float* m2_w    = (const float*)d_in[10];
  const float* m2_b    = (const float*)d_in[11];
  const float* psi_w   = (const float*)d_in[12];
  const float* psi_b   = (const float*)d_in[13];
  const float* z_w     = (const float*)d_in[14];
  const float* z_b     = (const float*)d_in[15];

  float* FI = (float*)d_ws;                       //  32768 x 64
  float* AI = FI + (size_t)NPTS * HD;             //  32768
  float* Ub = AI + NPTS;                          //  32768 x 128
  float* Sb = Ub + (size_t)NPTS * CC;             //  32768   (~25 MB total)

  k_prep<<<NPTS / 64, 128, 0, stream>>>(cxyz, cf, phi_w, phi_b, m1_w, m1_b,
                                        psi_w, FI, AI);
  k_main<<<NPTS / (WPB * PPW), 128, 0, stream>>>(cxyz, nxyz, nf, theta_w,
                                                 theta_b, m2_w, m2_b, psi_w,
                                                 psi_b, FI, AI, Ub, Sb);
  k_out<<<NPTS / 64, 128, 0, stream>>>(Ub, Sb, z_w, z_b, (float*)d_out);
}
